// SigmaFusion_48962627174676
// MI455X (gfx1250) — matrix operand order repaired and checked by reading, not compiled, not run on hardware
//
#include <hip/hip_runtime.h>
#include <hip/hip_bf16.h>

// ---- problem constants (match reference) ----
#define BB     4
#define LL     1024
#define DMODEL 512
#define DI     512          // d_inner
#define DS     16           // d_state
#define DTR    32           // dt_rank
#define NTOK   (BB*LL)      // 4096 tokens
#define FELEMS ((size_t)NTOK*DMODEL) // 2,097,152

// chunked scan decomposition
#define NC     32           // chunks along L
#define CL     (LL/NC)      // 32 steps per chunk
#define SEG    (BB*DI*DS)   // 32768 per-chunk state-vector length

typedef __attribute__((ext_vector_type(16))) __bf16 v16bf;
typedef __attribute__((ext_vector_type(8)))  float  v8f;
typedef unsigned short u16;

// float -> bf16 round-to-nearest-even
__device__ __forceinline__ u16 f2bf(float f) {
  union { float f; unsigned u; } x; x.f = f;
  unsigned u = x.u;
  if ((u & 0x7fffffffu) > 0x7f800000u) return (u16)((u >> 16) | 0x40u); // quiet NaN
  return (u16)((u + 0x7fffu + ((u >> 16) & 1u)) >> 16);
}

// per-lane WMMA bf16 operand: two 16B chunks at k offsets 0 and 16 (ISA 16-bit A/B layout)
__device__ __forceinline__ v16bf ldb16(const u16* p) {
  union { float4 f[2]; v16bf v; } u;
  u.f[0] = *reinterpret_cast<const float4*>(p);
  u.f[1] = *reinterpret_cast<const float4*>(p + 16);
  return u.v;
}

__device__ __forceinline__ float softplusf(float x) {
  return (x > 20.f) ? x : log1pf(__expf(x));
}
__device__ __forceinline__ float siluf(float x) {
  return x / (1.f + __expf(-x));
}

// =====================================================================
// GEMM: C[M,N](f32) = A[M,K](bf16, row-major, lda) * W[N,K](bf16, row-major, ldw)^T
// 64x64 block tile, 8 waves, each wave -> 16x32 via two v_wmma_f32_16x16x32_bf16.
// Optional bf16 mirror of C (for chaining into the next GEMM).
// =====================================================================
__global__ __launch_bounds__(256)
void gemm_wmma_bf16(const u16* __restrict__ A, int lda,
                    const u16* __restrict__ W, int ldw,
                    float* __restrict__ C, u16* __restrict__ Cbf,
                    int N, int K) {
  const int lane = threadIdx.x & 31;
  const int wave = threadIdx.x >> 5;
  const int m0 = blockIdx.y * 64 + (wave >> 1) * 16;
  const int n0 = blockIdx.x * 64 + (wave & 1) * 32;
  const int lm = lane & 15, hi = lane >> 4;

  const u16* a  = A + (size_t)(m0 + lm)      * lda + hi * 8;
  const u16* b0 = W + (size_t)(n0 + lm)      * ldw + hi * 8;
  const u16* b1 = W + (size_t)(n0 + 16 + lm) * ldw + hi * 8;

  v8f c0 = {}; v8f c1 = {};
  for (int k = 0; k < K; k += 32) {
    v16bf av  = ldb16(a  + k);
    v16bf bv0 = ldb16(b0 + k);
    v16bf bv1 = ldb16(b1 + k);
    c0 = __builtin_amdgcn_wmma_f32_16x16x32_bf16(false, av, false, bv0, (short)0, c0, false, false);
    c1 = __builtin_amdgcn_wmma_f32_16x16x32_bf16(false, av, false, bv1, (short)0, c1, false, false);
  }
  // C/D layout: VGPR r -> row m0 + 8*hi + r, col n0 + lm  (coalesced across lanes)
  float* crow = C + (size_t)(m0 + hi * 8) * N + n0 + lm;
  #pragma unroll
  for (int r = 0; r < 8; ++r) {
    crow[(size_t)r * N]      = c0[r];
    crow[(size_t)r * N + 16] = c1[r];
  }
  if (Cbf) {
    u16* brow = Cbf + (size_t)(m0 + hi * 8) * N + n0 + lm;
    #pragma unroll
    for (int r = 0; r < 8; ++r) {
      brow[(size_t)r * N]      = f2bf(c0[r]);
      brow[(size_t)r * N + 16] = f2bf(c1[r]);
    }
  }
}

// =====================================================================
// sub = LayerNorm(vi - ir) -> bf16 (GEMM A operand). One block per token.
// =====================================================================
__global__ __launch_bounds__(256)
void subln_kernel(const float* __restrict__ vi, const float* __restrict__ ir,
                  const float* __restrict__ w, const float* __restrict__ b,
                  u16* __restrict__ out) {
  const int tok = blockIdx.x, t = threadIdx.x;
  const size_t base = (size_t)tok * DMODEL;
  float x0 = vi[base + t]       - ir[base + t];
  float x1 = vi[base + t + 256] - ir[base + t + 256];
  __shared__ float red[256];
  red[t] = x0 + x1; __syncthreads();
  #pragma unroll
  for (int s = 128; s > 0; s >>= 1) { if (t < s) red[t] += red[t + s]; __syncthreads(); }
  float mu = red[0] * (1.f / DMODEL); __syncthreads();
  float d0 = x0 - mu, d1 = x1 - mu;
  red[t] = d0 * d0 + d1 * d1; __syncthreads();
  #pragma unroll
  for (int s = 128; s > 0; s >>= 1) { if (t < s) red[t] += red[t + s]; __syncthreads(); }
  float rs = rsqrtf(red[0] * (1.f / DMODEL) + 1e-5f);
  out[base + t]       = f2bf(d0 * rs * w[t]       + b[t]);
  out[base + t + 256] = f2bf(d1 * rs * w[t + 256] + b[t + 256]);
}

// =====================================================================
// depthwise causal conv(k=4) + SiLU, one thread per output element.
// dir=0 forward; dir=1 over reversed seq (output indexed by reversed time t).
// xz token-major (tok, 2*DI); x half = cols [0,DI). Coalesced across d.
// =====================================================================
__global__ __launch_bounds__(256)
void conv_silu_kernel(const float* __restrict__ xz,
                      const float* __restrict__ cw, const float* __restrict__ cb,
                      float* __restrict__ xo, u16* __restrict__ xo_bf, int dir) {
  const size_t i = (size_t)blockIdx.x * 256 + threadIdx.x; // = (b*LL + t)*DI + d
  const int d = (int)(i & (DI - 1));
  const int t = (int)((i >> 9) & (LL - 1));
  const int b = (int)(i >> 19);
  const size_t base = (size_t)b * LL;
  float acc = cb[d];
  #pragma unroll
  for (int j = 0; j < 4; ++j) {
    int s = dir ? (LL - 1 - t) + 3 - j : t - 3 + j; // original-sequence index for tap j
    if (s >= 0 && s < LL) acc += cw[d * 4 + j] * xz[(base + s) * (2 * DI) + d];
  }
  float v = siluf(acc);
  xo[i] = v; xo_bf[i] = f2bf(v);
}

// =====================================================================
// Chunked selective scan, exact linear-recurrence decomposition:
//   h_t = dA_t * h_{t-1} + dt_t*u_t*B_t   composes over a chunk as
//   h_end = (prod dA) * h_init + h_local.
// Phase 1: per-chunk local scan from h=0 -> store prod(dA) and h_local.
// Phase 2: 32-step sequential combine per (b,d,n) -> per-chunk h_init (in place).
// Phase 3: re-run chunk scan seeded with h_init, emit y (+u*D, *silu(z)).
// Inputs u/dtraw/xdbl are already direction-indexed; dir only affects phase 3
// output position and z lookup.
// =====================================================================
__global__ __launch_bounds__(256)
void scan_phase1(const float* __restrict__ u, const float* __restrict__ dtraw,
                 const float* __restrict__ xdbl,
                 const float* __restrict__ Alog, const float* __restrict__ dtb,
                 float* __restrict__ Ap, float* __restrict__ Hl) {
  const int d = blockIdx.x * 256 + threadIdx.x;  // grid.x = 2
  const int c = blockIdx.y;                      // chunk
  const int b = blockIdx.z;
  float A[DS], h[DS], ap[DS];
  #pragma unroll
  for (int n = 0; n < DS; ++n) { A[n] = -__expf(Alog[d * DS + n]); h[n] = 0.f; ap[n] = 1.f; }
  const float bias = dtb[d];
  const size_t base = (size_t)b * LL + (size_t)c * CL;
  for (int t = 0; t < CL; ++t) {
    const size_t tok = base + t;
    float uv = u[tok * DI + d];
    float dv = softplusf(dtraw[tok * DI + d] + bias);
    const float4* bc = reinterpret_cast<const float4*>(xdbl + tok * (DTR + 2 * DS) + DTR);
    float4 Bq[4];
    #pragma unroll
    for (int q = 0; q < 4; ++q) Bq[q] = bc[q];
    const float* Bn = reinterpret_cast<const float*>(Bq);
    float du = dv * uv;
    #pragma unroll
    for (int n = 0; n < DS; ++n) {
      float dA = __expf(dv * A[n]);
      ap[n] *= dA;
      h[n] = dA * h[n] + du * Bn[n];
    }
  }
  const size_t o = (size_t)c * SEG + ((size_t)b * DI + d) * DS;
  #pragma unroll
  for (int n = 0; n < DS; ++n) { Ap[o + n] = ap[n]; Hl[o + n] = h[n]; }
}

__global__ __launch_bounds__(256)
void scan_phase2(const float* __restrict__ Ap, float* __restrict__ Hl) {
  const int i = blockIdx.x * 256 + threadIdx.x;  // SEG threads
  float H = 0.f;
  for (int c = 0; c < NC; ++c) {
    const size_t idx = (size_t)c * SEG + i;
    float ap = Ap[idx];
    float hl = Hl[idx];
    Hl[idx] = H;              // h_init for chunk c
    H = ap * H + hl;          // carry to next chunk
  }
}

__global__ __launch_bounds__(256)
void scan_phase3(const float* __restrict__ u, const float* __restrict__ dtraw,
                 const float* __restrict__ xdbl, const float* __restrict__ xz,
                 const float* __restrict__ Alog, const float* __restrict__ dtb,
                 const float* __restrict__ Dv, const float* __restrict__ Hinit,
                 float* __restrict__ y, int dir) {
  const int d = blockIdx.x * 256 + threadIdx.x;
  const int c = blockIdx.y;
  const int b = blockIdx.z;
  float A[DS], h[DS];
  const size_t o = (size_t)c * SEG + ((size_t)b * DI + d) * DS;
  #pragma unroll
  for (int n = 0; n < DS; ++n) { A[n] = -__expf(Alog[d * DS + n]); h[n] = Hinit[o + n]; }
  const float Dd = Dv[d], bias = dtb[d];
  const size_t base = (size_t)b * LL;
  for (int t = 0; t < CL; ++t) {
    const int tg = c * CL + t;                 // global (direction-local) time
    const size_t tok = base + tg;
    const int lo = dir ? (LL - 1 - tg) : tg;   // original position
    const size_t tokO = base + lo;
    float uv = u[tok * DI + d];
    float dv = softplusf(dtraw[tok * DI + d] + bias);
    float zv = xz[tokO * (2 * DI) + DI + d];
    const float4* bc = reinterpret_cast<const float4*>(xdbl + tok * (DTR + 2 * DS) + DTR);
    float4 Bq[4], Cq[4];
    #pragma unroll
    for (int q = 0; q < 4; ++q) { Bq[q] = bc[q]; Cq[q] = bc[q + 4]; }
    const float* Bn = reinterpret_cast<const float*>(Bq);
    const float* Cn = reinterpret_cast<const float*>(Cq);
    float du = dv * uv, yv = 0.f;
    #pragma unroll
    for (int n = 0; n < DS; ++n) {
      float dA = __expf(dv * A[n]);
      h[n] = dA * h[n] + du * Bn[n];
      yv += h[n] * Cn[n];
    }
    yv += uv * Dd;
    y[tokO * DI + d] = yv * siluf(zv);
  }
}

__global__ __launch_bounds__(256)
void add_bf16_kernel(const float* __restrict__ a, const float* __restrict__ b,
                     u16* __restrict__ o) {
  size_t i = (size_t)blockIdx.x * 256 + threadIdx.x;
  o[i] = f2bf(a[i] + b[i]);
}

// g = tanh(feat); vi/ir cross-gate; build bf16 [vi|ir] concat rows for fuse GEMM
__global__ __launch_bounds__(256)
void gate_kernel(const float* __restrict__ feat, const float* __restrict__ vi0,
                 const float* __restrict__ ir0, float* __restrict__ vo,
                 float* __restrict__ io, u16* __restrict__ cat) {
  size_t i = (size_t)blockIdx.x * 256 + threadIdx.x;
  float g = tanhf(feat[i]);
  float a = vi0[i], c = ir0[i];
  float vi = a + g * c, ir = c + g * a;
  vo[i] = vi; io[i] = ir;
  size_t tok = i >> 9; int col = (int)(i & 511);
  cat[tok * 1024 + col]       = f2bf(vi);
  cat[tok * 1024 + 512 + col] = f2bf(ir);
}

// pre + fuse_b -> LayerNorm -> exact GELU -> d_out[0..F)
__global__ __launch_bounds__(256)
void fuse_ln_gelu_kernel(const float* __restrict__ pre, const float* __restrict__ fb,
                         const float* __restrict__ lw, const float* __restrict__ lb,
                         float* __restrict__ out) {
  const int tok = blockIdx.x, t = threadIdx.x;
  const size_t base = (size_t)tok * DMODEL;
  float x0 = pre[base + t]       + fb[t];
  float x1 = pre[base + t + 256] + fb[t + 256];
  __shared__ float red[256];
  red[t] = x0 + x1; __syncthreads();
  #pragma unroll
  for (int s = 128; s > 0; s >>= 1) { if (t < s) red[t] += red[t + s]; __syncthreads(); }
  float mu = red[0] * (1.f / DMODEL); __syncthreads();
  float d0 = x0 - mu, d1 = x1 - mu;
  red[t] = d0 * d0 + d1 * d1; __syncthreads();
  #pragma unroll
  for (int s = 128; s > 0; s >>= 1) { if (t < s) red[t] += red[t + s]; __syncthreads(); }
  float rs = rsqrtf(red[0] * (1.f / DMODEL) + 1e-5f);
  float y0 = d0 * rs * lw[t]       + lb[t];
  float y1 = d1 * rs * lw[t + 256] + lb[t + 256];
  out[base + t]       = 0.5f * y0 * (1.f + erff(y0 * 0.70710678118654752f));
  out[base + t + 256] = 0.5f * y1 * (1.f + erff(y1 * 0.70710678118654752f));
}

__global__ __launch_bounds__(256)
void cvt_bf16_kernel(const float* __restrict__ in, u16* __restrict__ out, int n) {
  int i = blockIdx.x * 256 + threadIdx.x;
  if (i < n) out[i] = f2bf(in[i]);
}

extern "C" void kernel_launch(void* const* d_in, const int* in_sizes, int n_in,
                              void* d_out, int out_size, void* d_ws, size_t ws_size,
                              hipStream_t stream) {
  (void)in_sizes; (void)n_in; (void)out_size; (void)ws_size;
  const float* oral_vi    = (const float*)d_in[0];
  const float* oral_ir    = (const float*)d_in[1];
  const float* norm_w     = (const float*)d_in[2];
  const float* norm_b     = (const float*)d_in[3];
  const float* in_proj_w  = (const float*)d_in[4];
  const float* conv_w     = (const float*)d_in[5];
  const float* conv_b     = (const float*)d_in[6];
  const float* x_proj_w   = (const float*)d_in[7];
  const float* dt_proj_w  = (const float*)d_in[8];
  const float* dt_proj_b  = (const float*)d_in[9];
  const float* A_log      = (const float*)d_in[10];
  const float* A_b_log    = (const float*)d_in[11];
  const float* Dvec       = (const float*)d_in[12];
  const float* out_proj_w = (const float*)d_in[13];
  const float* fuse_w     = (const float*)d_in[14];
  const float* fuse_b     = (const float*)d_in[15];
  const float* fuse_ln_w  = (const float*)d_in[16];
  const float* fuse_ln_b  = (const float*)d_in[17];

  // ---- workspace bump layout (~86 MB), with aliasing of dead buffers ----
  char* w = (char*)d_ws;
  float* xz       = (float*)(w);                     // 16 MB (tok, 1024); dead after scans
  float* x_f      = (float*)(w + (16ull << 20));     // 8 MB; dead after scans
  float* x_b      = (float*)(w + (24ull << 20));     // 8 MB
  float* dtraw_f  = (float*)(w + (32ull << 20));     // 8 MB; dead after scans
  float* dtraw_b  = (float*)(w + (40ull << 20));     // 8 MB
  float* y_f      = (float*)(w + (48ull << 20));     // 8 MB
  float* y_b      = (float*)(w + (56ull << 20));     // 8 MB
  u16*   subA     = (u16*)  (w + (64ull << 20));     // 4 MB; dead after in_proj GEMM
  u16*   xf_bf    = (u16*)  (w + (68ull << 20));     // 4 MB; dead after x_proj GEMM
  u16*   xb_bf    = (u16*)  (w + (72ull << 20));     // 4 MB; dead after x_proj GEMM
  u16*   ys_bf    = (u16*)  (w + (76ull << 20));     // 4 MB; written after scans
  float* xdbl_f   = (float*)(w + (80ull << 20));     // 1 MB (tok, 64)
  float* xdbl_b   = (float*)(w + (81ull << 20));     // 1 MB
  u16*   wip      = (u16*)  (w + (82ull << 20));     // 1 MB   in_proj bf16
  u16*   wfu      = (u16*)  (w + (83ull << 20));     // 1 MB   fuse bf16
  u16*   wop      = (u16*)  (w + (84ull << 20));                  // 512 KB out_proj
  u16*   wxp      = (u16*)  (w + (84ull << 20) + (512ull << 10)); // 64 KB x_proj
  u16*   wdt      = (u16*)  (w + (84ull << 20) + (576ull << 10)); // 32 KB dt_proj
  u16*   xdblf_bf = (u16*)  (w + (84ull << 20) + (640ull << 10)); // 512 KB
  u16*   xdblb_bf = (u16*)  (w + (84ull << 20) + (1152ull << 10));// 512 KB

  // chunked-scan state buffers (4 MB each) alias bf16 buffers dead by scan time
  float* Ap_f = (float*)subA;    // prod(dA), forward
  float* Hl_f = (float*)xf_bf;   // local h -> h_init, forward
  float* Ap_b = (float*)xb_bf;   // prod(dA), backward
  float* Hl_b = (float*)ys_bf;   // local h -> h_init, backward (ys_bf written later)

  float* feat   = x_f;             // alias: x_f dead after scans
  u16*   cat_bf = (u16*)dtraw_f;   // alias: dtraw_f dead after scans
  float* pre    = xz;              // alias: xz dead after scans

  float* out_fused = (float*)d_out;
  float* out_vi    = out_fused + FELEMS;
  float* out_ir    = out_vi + FELEMS;

  // weight f32 -> bf16 (small, once per call; deterministic)
  cvt_bf16_kernel<<<(1024 * 512 + 255) / 256, 256, 0, stream>>>(in_proj_w,  wip, 1024 * 512);
  cvt_bf16_kernel<<<(64 * 512 + 255) / 256,   256, 0, stream>>>(x_proj_w,   wxp, 64 * 512);
  cvt_bf16_kernel<<<(512 * 32 + 255) / 256,   256, 0, stream>>>(dt_proj_w,  wdt, 512 * 32);
  cvt_bf16_kernel<<<(512 * 512 + 255) / 256,  256, 0, stream>>>(out_proj_w, wop, 512 * 512);
  cvt_bf16_kernel<<<(512 * 1024 + 255) / 256, 256, 0, stream>>>(fuse_w,     wfu, 512 * 1024);

  // 1) sub = LN(vi - ir) -> bf16
  subln_kernel<<<NTOK, 256, 0, stream>>>(oral_vi, oral_ir, norm_w, norm_b, subA);

  // 2) xz = sub @ in_proj_w^T : (4096x512)x(512x1024)
  gemm_wmma_bf16<<<dim3(1024 / 64, NTOK / 64), 256, 0, stream>>>(subA, 512, wip, 512, xz, nullptr, 1024, 512);

  // 3) depthwise causal conv + SiLU, both directions (1 thread / element)
  conv_silu_kernel<<<(int)(FELEMS / 256), 256, 0, stream>>>(xz, conv_w, conv_b, x_f, xf_bf, 0);
  conv_silu_kernel<<<(int)(FELEMS / 256), 256, 0, stream>>>(xz, conv_w, conv_b, x_b, xb_bf, 1);

  // 4) x_dbl = x @ x_proj_w^T : (4096x512)x(512x64); f32 for scan B/C, bf16 for dt GEMM
  gemm_wmma_bf16<<<dim3(1, NTOK / 64), 256, 0, stream>>>(xf_bf, 512, wxp, 512, xdbl_f, xdblf_bf, 64, 512);
  gemm_wmma_bf16<<<dim3(1, NTOK / 64), 256, 0, stream>>>(xb_bf, 512, wxp, 512, xdbl_b, xdblb_bf, 64, 512);

  // 5) dt_raw = x_dbl[:, :32] @ dt_proj_w^T : (4096x32)x(32x512)  (lda=64 selects dt cols)
  gemm_wmma_bf16<<<dim3(512 / 64, NTOK / 64), 256, 0, stream>>>(xdblf_bf, 64, wdt, 32, dtraw_f, nullptr, 512, 32);
  gemm_wmma_bf16<<<dim3(512 / 64, NTOK / 64), 256, 0, stream>>>(xdblb_bf, 64, wdt, 32, dtraw_b, nullptr, 512, 32);

  // 6) chunked selective scans (critical path 1024 -> ~96 dependent steps)
  scan_phase1<<<dim3(2, NC, BB), 256, 0, stream>>>(x_f, dtraw_f, xdbl_f, A_log,   dt_proj_b, Ap_f, Hl_f);
  scan_phase1<<<dim3(2, NC, BB), 256, 0, stream>>>(x_b, dtraw_b, xdbl_b, A_b_log, dt_proj_b, Ap_b, Hl_b);
  scan_phase2<<<SEG / 256, 256, 0, stream>>>(Ap_f, Hl_f);
  scan_phase2<<<SEG / 256, 256, 0, stream>>>(Ap_b, Hl_b);
  scan_phase3<<<dim3(2, NC, BB), 256, 0, stream>>>(x_f, dtraw_f, xdbl_f, xz, A_log,   dt_proj_b, Dvec, Hl_f, y_f, 0);
  scan_phase3<<<dim3(2, NC, BB), 256, 0, stream>>>(x_b, dtraw_b, xdbl_b, xz, A_b_log, dt_proj_b, Dvec, Hl_b, y_b, 1);

  // 7) y = y_f + y_b -> bf16 (ys_bf aliases Hl_b region; scans are complete)
  add_bf16_kernel<<<(int)(FELEMS / 256), 256, 0, stream>>>(y_f, y_b, ys_bf);

  // 8) feat = y @ out_proj_w^T : (4096x512)x(512x512)
  gemm_wmma_bf16<<<dim3(512 / 64, NTOK / 64), 256, 0, stream>>>(ys_bf, 512, wop, 512, feat, nullptr, 512, 512);

  // 9) tanh gate, write vi/ir outputs, build concat rows
  gate_kernel<<<(int)(FELEMS / 256), 256, 0, stream>>>(feat, oral_vi, oral_ir, out_vi, out_ir, cat_bf);

  // 10) pre = cat @ fuse_w^T : (4096x1024)x(1024x512)
  gemm_wmma_bf16<<<dim3(512 / 64, NTOK / 64), 256, 0, stream>>>(cat_bf, 1024, wfu, 1024, pre, nullptr, 512, 1024);

  // 11) + bias, LayerNorm, exact GELU -> fused output
  fuse_ln_gelu_kernel<<<NTOK, 256, 0, stream>>>(pre, fuse_b, fuse_ln_w, fuse_ln_b, out_fused);
}
